// TransformerSelfAttentionRing_49417893707801
// MI455X (gfx1250) — matrix operand order repaired and checked by reading, not compiled, not run on hardware
//
#include <hip/hip_runtime.h>
#include <hip/hip_bf16.h>

#define S_LEN 2048
#define B_SZ  2
#define H_SZ  2048
#define NH_SZ 16
#define DH_SZ 128
#define QKV3H 6144

typedef __bf16 bf16_t;
typedef __attribute__((ext_vector_type(16))) bf16_t v16bf;
typedef __attribute__((ext_vector_type(8)))  float  v8f;

union FragAB {
  v16bf v;
  uint4 q[2];
  unsigned int u[8];
  unsigned short h[16];
};

__device__ __forceinline__ unsigned short f2bf(float f) {
  unsigned int u = __float_as_uint(f);
  u += 0x7FFFu + ((u >> 16) & 1u);   // round-to-nearest-even
  return (unsigned short)(u >> 16);
}

// ---------------------------------------------------------------------------
// Tiled bf16 WMMA GEMM: D[M,N] = A[M,K] * B[K,N] (+bias), fp32 accumulate.
// AT: float (converted on the fly) or unsigned short (raw bf16).
// Block 256 threads (8 wave32), tile 128x128, K-step 32.
// ---------------------------------------------------------------------------
template <typename AT, bool OUT_BF16>
__global__ __launch_bounds__(256)
void gemm_wmma(const AT* __restrict__ A, const float* __restrict__ Bw,
               const float* __restrict__ bias, void* __restrict__ Dst,
               int M, int N, int K) {
  __shared__ unsigned short sA[128 * 40];   // [m][k], padded row stride 40
  __shared__ unsigned short sB[128 * 40];   // transposed: [n][k], stride 40

  const int tid  = threadIdx.x;
  const int lane = tid & 31;
  const int wave = tid >> 5;
  const int g    = (lane >> 4) & 1;
  const int ln   = lane & 15;

  const int waveM = wave >> 1;   // 0..3  -> 32 rows each
  const int waveN = wave & 1;    // 0..1  -> 64 cols each
  const int row0  = blockIdx.y * 128;
  const int col0  = blockIdx.x * 128;

  v8f acc[2][4];
  for (int i = 0; i < 2; ++i)
    for (int j = 0; j < 4; ++j)
      for (int r = 0; r < 8; ++r) acc[i][j][r] = 0.0f;

  const int nK = K >> 5;
  for (int kb = 0; kb < nK; ++kb) {
    const int kk = kb << 5;
    __syncthreads();

    if (kb + 1 < nK) {  // global_prefetch_b8 of next K tile
      __builtin_prefetch(Bw + (size_t)(kk + 32 + (tid >> 3)) * N + col0, 0, 0);
      __builtin_prefetch(A + (size_t)(row0 + (tid >> 1)) * K + kk + 32, 0, 0);
    }

    // ---- stage A tile (128 x 32) as bf16 ----
    if constexpr (sizeof(AT) == 4) {
      for (int i = 0; i < 4; ++i) {
        int idx = tid + i * 256;              // 1024 float4
        int r = idx >> 3, c = idx & 7;
        const float4 f = *reinterpret_cast<const float4*>(
            (const float*)A + (size_t)(row0 + r) * K + kk + c * 4);
        unsigned int p0 = (unsigned int)f2bf(f.x) | ((unsigned int)f2bf(f.y) << 16);
        unsigned int p1 = (unsigned int)f2bf(f.z) | ((unsigned int)f2bf(f.w) << 16);
        unsigned int* d = reinterpret_cast<unsigned int*>(&sA[r * 40 + c * 4]);
        d[0] = p0; d[1] = p1;
      }
    } else {
      for (int i = 0; i < 8; ++i) {
        int idx = tid + i * 256;              // 2048 dwords
        int r = idx >> 4, c = idx & 15;
        unsigned int wv = *reinterpret_cast<const unsigned int*>(
            (const unsigned short*)A + (size_t)(row0 + r) * K + kk + c * 2);
        *reinterpret_cast<unsigned int*>(&sA[r * 40 + c * 2]) = wv;
      }
    }
    // ---- stage B tile (32 x 128) transposed -> sB[n][k] ----
    for (int i = 0; i < 4; ++i) {
      int idx = tid + i * 256;                // 1024 float4
      int r = idx >> 5, c4 = idx & 31;
      const float4 f = *reinterpret_cast<const float4*>(
          Bw + (size_t)(kk + r) * N + col0 + c4 * 4);
      sB[(c4 * 4 + 0) * 40 + r] = f2bf(f.x);
      sB[(c4 * 4 + 1) * 40 + r] = f2bf(f.y);
      sB[(c4 * 4 + 2) * 40 + r] = f2bf(f.z);
      sB[(c4 * 4 + 3) * 40 + r] = f2bf(f.w);
    }
    __syncthreads();

    // ---- fragments ----
    FragAB af[2], bfr[4];
    for (int mi = 0; mi < 2; ++mi) {
      const unsigned short* base = &sA[(waveM * 32 + mi * 16 + ln) * 40];
      for (int j = 0; j < 4; ++j)
        af[mi].u[j] = *reinterpret_cast<const unsigned int*>(base + 8 * g + 2 * j);
      for (int j = 0; j < 4; ++j)
        af[mi].u[4 + j] = *reinterpret_cast<const unsigned int*>(base + 16 + 8 * g + 2 * j);
    }
    for (int ni = 0; ni < 4; ++ni) {
      const uint4* p = reinterpret_cast<const uint4*>(
          &sB[(waveN * 64 + ni * 16 + ln) * 40 + 8 * g]);
      bfr[ni].q[0] = p[0];
      bfr[ni].q[1] = p[2];
    }
    for (int mi = 0; mi < 2; ++mi)
      for (int ni = 0; ni < 4; ++ni)
        acc[mi][ni] = __builtin_amdgcn_wmma_f32_16x16x32_bf16(
            false, af[mi].v, false, bfr[ni].v, (short)0, acc[mi][ni], false, false);
  }

  // ---- epilogue ----
  for (int mi = 0; mi < 2; ++mi) {
    for (int ni = 0; ni < 4; ++ni) {
      const int col = col0 + waveN * 64 + ni * 16 + ln;
      const float bv = bias ? bias[col] : 0.0f;
      for (int r = 0; r < 8; ++r) {
        const int row = row0 + waveM * 32 + mi * 16 + 8 * g + r;
        const float val = acc[mi][ni][r] + bv;
        if constexpr (OUT_BF16)
          reinterpret_cast<unsigned short*>(Dst)[(size_t)row * N + col] = f2bf(val);
        else
          reinterpret_cast<float*>(Dst)[(size_t)row * N + col] = val;
      }
    }
  }
}

// ---------------------------------------------------------------------------
// Flash attention: grid (S/128, NH, B), 8 waves/block, 16 query rows per wave.
// Online softmax over 32-key blocks, all matmuls on v_wmma_f32_16x16x32_bf16.
// ---------------------------------------------------------------------------
__global__ __launch_bounds__(256)
void attn_flash(const unsigned short* __restrict__ mixed,   // bf16 (S*B, 6144)
                const unsigned char* __restrict__ mask,     // (B,1,S,S) bool
                unsigned short* __restrict__ ctx) {         // bf16 (S*B, 2048)
  __shared__ unsigned short Klds[32 * 136];    // [key][d]
  __shared__ unsigned short Vlds[128 * 40];    // transposed: [d][key]
  __shared__ unsigned short Plds[8 * 16 * 40]; // per-wave P scratch

  const int tid  = threadIdx.x;
  const int lane = tid & 31;
  const int wave = tid >> 5;
  const int g    = (lane >> 4) & 1;
  const int ln   = lane & 15;

  const int head = blockIdx.y;
  const int b    = blockIdx.z;
  const int qbase = blockIdx.x * 128 + wave * 16;

  const size_t qcol = (size_t)head * 384;
  const size_t kcol = qcol + 128;
  const size_t vcol = qcol + 256;

  // preload Q fragments (A-layout) for 4 k-steps over DH=128
  FragAB qf[4];
  {
    const size_t rowoff = ((size_t)(qbase + ln) * B_SZ + b) * QKV3H + qcol;
    for (int st = 0; st < 4; ++st) {
      for (int j = 0; j < 4; ++j)
        qf[st].u[j] = *reinterpret_cast<const unsigned int*>(
            mixed + rowoff + st * 32 + 8 * g + 2 * j);
      for (int j = 0; j < 4; ++j)
        qf[st].u[4 + j] = *reinterpret_cast<const unsigned int*>(
            mixed + rowoff + st * 32 + 16 + 8 * g + 2 * j);
    }
  }

  v8f acc[8];
  for (int t = 0; t < 8; ++t)
    for (int r = 0; r < 8; ++r) acc[t][r] = 0.0f;
  float mrow[8], lrow[8];
  for (int r = 0; r < 8; ++r) { mrow[r] = -3.0e38f; lrow[r] = 0.0f; }

  const float inv_norm = 0.022097086912079608f;  // 1/sqrt(2048)
  unsigned short* myP = &Plds[wave * 16 * 40];

  for (int kb = 0; kb < S_LEN / 32; ++kb) {
    const int kk = kb * 32;
    __syncthreads();
    // stage K tile (32 x 128) row-major
    for (int i = 0; i < 8; ++i) {
      int idx = tid + i * 256;
      int key = idx >> 6, dw = idx & 63;
      unsigned int wv = *reinterpret_cast<const unsigned int*>(
          mixed + ((size_t)(kk + key) * B_SZ + b) * QKV3H + kcol + dw * 2);
      *reinterpret_cast<unsigned int*>(&Klds[key * 136 + dw * 2]) = wv;
    }
    // stage V tile (32 x 128) transposed -> Vlds[d][key]
    for (int i = 0; i < 8; ++i) {
      int idx = tid + i * 256;
      int key = idx >> 6, dw = idx & 63;
      unsigned int wv = *reinterpret_cast<const unsigned int*>(
          mixed + ((size_t)(kk + key) * B_SZ + b) * QKV3H + vcol + dw * 2);
      Vlds[(dw * 2 + 0) * 40 + key] = (unsigned short)(wv & 0xFFFFu);
      Vlds[(dw * 2 + 1) * 40 + key] = (unsigned short)(wv >> 16);
    }
    __syncthreads();

    // scores: two 16x16 tiles (keys kk..kk+15, kk+16..kk+31)
    v8f sc[2];
    for (int t = 0; t < 2; ++t) {
      for (int r = 0; r < 8; ++r) sc[t][r] = 0.0f;
      const int key = t * 16 + ln;
      for (int st = 0; st < 4; ++st) {
        FragAB kf;
        const uint4* p = reinterpret_cast<const uint4*>(
            &Klds[key * 136 + st * 32 + 8 * g]);
        kf.q[0] = p[0]; kf.q[1] = p[2];
        sc[t] = __builtin_amdgcn_wmma_f32_16x16x32_bf16(
            false, qf[st].v, false, kf.v, (short)0, sc[t], false, false);
      }
    }

    // scale + mask (exact replace with -10000, like the reference)
    const int qr0 = qbase + 8 * g;
    for (int t = 0; t < 2; ++t) {
      const int key = kk + t * 16 + ln;
      for (int r = 0; r < 8; ++r) {
        float s = sc[t][r] * inv_norm;
        unsigned char mv = mask[((size_t)b * S_LEN + (qr0 + r)) * S_LEN + key];
        sc[t][r] = mv ? -10000.0f : s;
      }
    }

    // online softmax update (row index is per-register -> lane-local rescale)
    float p0[8], p1[8];
    for (int r = 0; r < 8; ++r) {
      float mx = fmaxf(sc[0][r], sc[1][r]);
      for (int off = 1; off < 16; off <<= 1)
        mx = fmaxf(mx, __shfl_xor(mx, off, 32));
      const float nm   = fmaxf(mrow[r], mx);
      const float corr = __expf(mrow[r] - nm);
      mrow[r] = nm;
      const float e0 = __expf(sc[0][r] - nm);
      const float e1 = __expf(sc[1][r] - nm);
      float sum = e0 + e1;
      for (int off = 1; off < 16; off <<= 1)
        sum += __shfl_xor(sum, off, 32);
      lrow[r] = lrow[r] * corr + sum;
      for (int t = 0; t < 8; ++t) acc[t][r] *= corr;
      p0[r] = e0; p1[r] = e1;
    }

    // C-layout -> A-layout via per-wave LDS patch
    for (int r = 0; r < 8; ++r) {
      myP[(8 * g + r) * 40 + ln]      = f2bf(p0[r]);
      myP[(8 * g + r) * 40 + 16 + ln] = f2bf(p1[r]);
    }
    FragAB pf;
    {
      const unsigned short* base = &myP[ln * 40];
      for (int j = 0; j < 4; ++j)
        pf.u[j] = *reinterpret_cast<const unsigned int*>(base + 8 * g + 2 * j);
      for (int j = 0; j < 4; ++j)
        pf.u[4 + j] = *reinterpret_cast<const unsigned int*>(base + 16 + 8 * g + 2 * j);
    }

    // acc += P (16x32) x V (32x128)
    for (int t = 0; t < 8; ++t) {
      FragAB vf;
      const uint4* p = reinterpret_cast<const uint4*>(
          &Vlds[(t * 16 + ln) * 40 + 8 * g]);
      vf.q[0] = p[0]; vf.q[1] = p[2];
      acc[t] = __builtin_amdgcn_wmma_f32_16x16x32_bf16(
          false, pf.v, false, vf.v, (short)0, acc[t], false, false);
    }
  }

  // normalize and write context (bf16)
  for (int t = 0; t < 8; ++t) {
    const int d = t * 16 + ln;
    for (int r = 0; r < 8; ++r) {
      const int row = qbase + 8 * g + r;
      const float val = acc[t][r] / lrow[r];
      ctx[((size_t)row * B_SZ + b) * H_SZ + head * DH_SZ + d] = f2bf(val);
    }
  }
}

__global__ void copy_bias(const float* __restrict__ src, float* __restrict__ dst, int n) {
  int i = blockIdx.x * blockDim.x + threadIdx.x;
  if (i < n) dst[i] = src[i];
}

extern "C" void kernel_launch(void* const* d_in, const int* in_sizes, int n_in,
                              void* d_out, int out_size, void* d_ws, size_t ws_size,
                              hipStream_t stream) {
  const float*         hidden  = (const float*)d_in[0];
  const unsigned char* mask    = (const unsigned char*)d_in[1];
  const float*         W_qkv   = (const float*)d_in[2];
  const float*         b_qkv   = (const float*)d_in[3];
  const float*         W_dense = (const float*)d_in[4];
  const float*         b_dense = (const float*)d_in[5];
  float*               out     = (float*)d_out;

  unsigned short* mixed = (unsigned short*)d_ws;            // bf16 (4096, 6144)
  unsigned short* ctx   = mixed + (size_t)4096 * QKV3H;     // bf16 (4096, 2048)

  // 1) QKV projection (+b_qkv), bf16 output
  gemm_wmma<float, true><<<dim3(QKV3H / 128, 4096 / 128), dim3(256), 0, stream>>>(
      hidden, W_qkv, b_qkv, (void*)mixed, 4096, QKV3H, 2048);

  // 2) flash attention per (query-block, head, batch)
  attn_flash<<<dim3(S_LEN / 128, NH_SZ, B_SZ), dim3(256), 0, stream>>>(
      mixed, mask, ctx);

  // 3) output dense (no bias added; reference returns b_dense separately)
  gemm_wmma<unsigned short, false><<<dim3(H_SZ / 128, 4096 / 128), dim3(256), 0, stream>>>(
      ctx, W_dense, nullptr, (void*)out, 4096, H_SZ, 2048);

  // 4) append b_dense to the output tuple tail
  copy_bias<<<dim3((H_SZ + 255) / 256), dim3(256), 0, stream>>>(
      b_dense, out + (size_t)4096 * H_SZ, H_SZ);
}
